// ScaledDotProductAttention_83107617177617
// MI455X (gfx1250) — matrix operand order repaired and checked by reading, not compiled, not run on hardware
//
#include <hip/hip_runtime.h>
#include <hip/hip_bf16.h>

// Flash-attention forward (causal), fp32 in/out, bf16 WMMA compute, f32 accum.
// B=4, H=16, S=2048, D=64.
// One wave handles a 32-row Q tile (two 16x16 WMMA M tiles); K processed in
// 32-column chunks -> 16 v_wmma_f32_16x16x32_bf16 per chunk, K/V operands
// reused across both M tiles.

#define BB 4
#define HH 16
#define SS 2048
#define DD 64

typedef __attribute__((ext_vector_type(16))) __bf16 v16bf;
typedef __attribute__((ext_vector_type(8)))  float  v8f;

__device__ __forceinline__ void wave_lds_sync() {
  // Wave-local ordering of LDS traffic (each wave owns its own LDS region and
  // its own causal trip count, so no cross-wave barrier is wanted here).
  __builtin_amdgcn_wave_barrier();
  asm volatile("s_wait_dscnt 0" ::: "memory");
  __builtin_amdgcn_wave_barrier();
}

__global__ __launch_bounds__(256, 1) void fa_fwd_causal_kernel(
    const float* __restrict__ Q, const float* __restrict__ K,
    const float* __restrict__ V, float* __restrict__ O)
{
  // Per-wave scratch: scores 32x32 f32, P 32x32 bf16, per-row stats.
  __shared__ float  sc[8][32][32];
  __shared__ __bf16 pb[8][32][32];
  __shared__ float  st[8][32];

  const int wid  = threadIdx.x >> 5;
  const int lane = threadIdx.x & 31;
  const int hh   = lane >> 4;   // half-wave: 0 or 1
  const int lrow = lane & 15;

  const int wtile = blockIdx.x * 8 + wid;   // 0..4095 wave tiles (32 q-rows each)
  const int bh    = wtile >> 6;             // 64 q-tiles per (b,h)
  const int qt    = wtile & 63;
  const int qbase = qt << 5;                // 32-row aligned

  const float* qp = Q + (size_t)bh * SS * DD;
  const float* kp = K + (size_t)bh * SS * DD;
  const float* vp = V + (size_t)bh * SS * DD;
  float*       op = O + (size_t)bh * SS * DD;

  // ---- Q tile as four 16x32 bf16 A operands: aq[mt][dseg] ----
  // A layout used here (contiguous k, not the ISA table order): lane (hh,lrow) holds M=lrow;
  // element e: K = hh*16 + e
  v16bf aq[2][2];
  #pragma unroll
  for (int mt = 0; mt < 2; ++mt) {
    const float* qr = qp + (size_t)(qbase + mt * 16 + lrow) * DD;
    #pragma unroll
    for (int e = 0; e < 8; ++e) {
      const int d0 = hh * 16 + e;        // K-dim hh*16 + 0..7
      const int d1 = hh * 16 + 8 + e;   // K-dim hh*16 + 8..15
      aq[mt][0][e]     = (__bf16)qr[d0];
      aq[mt][0][e + 8] = (__bf16)qr[d1];
      aq[mt][1][e]     = (__bf16)qr[32 + d0];
      aq[mt][1][e + 8] = (__bf16)qr[32 + d1];
    }
  }

  v8f acc[2][4];
  #pragma unroll
  for (int mt = 0; mt < 2; ++mt)
    #pragma unroll
    for (int t = 0; t < 4; ++t)
      acc[mt][t] = (v8f){};

  float m_i = -__builtin_inff();  // running max for row `lane`
  float l_i = 0.0f;               // running sum for row `lane`
  const float scale = 0.125f;     // 1/sqrt(64)

  const int nchunk = (qbase >> 5) + 1;   // 32-column K chunks (causal)
  for (int kc = 0; kc < nchunk; ++kc) {
    const int  kb   = kc << 5;
    const bool full = (kb + 31) <= qbase;  // only the diagonal chunk is partial

    if (kc + 1 < nchunk)
      __builtin_prefetch(kp + (size_t)(kb + 32) * DD, 0, 1);

    // ---- K as four 32x16 bf16 B operands: bk[nt][dseg] ----
    // B layout: lane (hh,lrow): N=lrow, element e: K-dim = hh*16 + e
    v16bf bk[2][2];
    #pragma unroll
    for (int nt = 0; nt < 2; ++nt) {
      const float* kr = kp + (size_t)(kb + nt * 16 + lrow) * DD + hh * 16;
      #pragma unroll
      for (int e = 0; e < 16; ++e) {
        bk[nt][0][e] = (__bf16)kr[e];
        bk[nt][1][e] = (__bf16)kr[32 + e];
      }
    }

    // ---- QK^T: 8 WMMAs -> four 16x16 f32 score tiles ----
    v8f s[2][2];
    #pragma unroll
    for (int mt = 0; mt < 2; ++mt)
      #pragma unroll
      for (int nt = 0; nt < 2; ++nt) {
        v8f t0 = (v8f){};
        t0 = __builtin_amdgcn_wmma_f32_16x16x32_bf16(false, aq[mt][0], false, bk[nt][0], (short)0, t0, false, false);
        t0 = __builtin_amdgcn_wmma_f32_16x16x32_bf16(false, aq[mt][1], false, bk[nt][1], (short)0, t0, false, false);
        s[mt][nt] = t0;
      }

    // Scale + causal mask, stash scores in LDS.
    // C layout: VGPR r -> row = mt*16 + hh*8 + r, col = nt*16 + lrow
    #pragma unroll
    for (int mt = 0; mt < 2; ++mt)
      #pragma unroll
      for (int nt = 0; nt < 2; ++nt)
        #pragma unroll
        for (int r = 0; r < 8; ++r) {
          const int row = qbase + mt * 16 + hh * 8 + r;
          const int col = kb + nt * 16 + lrow;
          float v0 = s[mt][nt][r] * scale;
          if (!full && col > row) v0 = -__builtin_inff();
          sc[wid][mt * 16 + hh * 8 + r][nt * 16 + lrow] = v0;
        }
    wave_lds_sync();

    // ---- online softmax: lane owns row `lane` (all 32 lanes distinct) ----
    float cmax = -__builtin_inff();
    #pragma unroll
    for (int c = 0; c < 32; ++c) cmax = fmaxf(cmax, sc[wid][lane][c]);
    const float m_new = fmaxf(m_i, cmax);
    const float alpha = __expf(m_i - m_new);   // 0 on first chunk (m_i=-inf)
    float psum = 0.0f;
    #pragma unroll
    for (int c = 0; c < 32; ++c) {
      const float p = __expf(sc[wid][lane][c] - m_new);  // -inf -> 0
      psum += p;
      pb[wid][lane][c] = (__bf16)p;
    }
    l_i = l_i * alpha + psum;
    m_i = m_new;
    st[wid][lane] = alpha;
    wave_lds_sync();

    // ---- rescale output accumulators by per-row alpha ----
    #pragma unroll
    for (int mt = 0; mt < 2; ++mt)
      #pragma unroll
      for (int r = 0; r < 8; ++r) {
        const float a = st[wid][mt * 16 + hh * 8 + r];
        #pragma unroll
        for (int t = 0; t < 4; ++t) acc[mt][t][r] *= a;
      }

    // ---- P back as two 16x32 bf16 A operands ----
    v16bf ap[2];
    #pragma unroll
    for (int mt = 0; mt < 2; ++mt)
      #pragma unroll
      for (int e = 0; e < 8; ++e) {
        ap[mt][e]     = pb[wid][mt * 16 + lrow][hh * 16 + e];
        ap[mt][e + 8] = pb[wid][mt * 16 + lrow][hh * 16 + 8 + e];
      }

    // ---- V chunk (32 x 64) as four 32x16 B operands (coalesced per e) ----
    v16bf bv[4];
    #pragma unroll
    for (int e = 0; e < 16; ++e) {
      const float* vr = vp + (size_t)(kb + hh * 16 + e) * DD + lrow;
      #pragma unroll
      for (int t = 0; t < 4; ++t) bv[t][e] = (__bf16)vr[t * 16];
    }

    // ---- PV: 8 WMMAs, V operands shared across both M tiles ----
    #pragma unroll
    for (int mt = 0; mt < 2; ++mt)
      #pragma unroll
      for (int t = 0; t < 4; ++t)
        acc[mt][t] = __builtin_amdgcn_wmma_f32_16x16x32_bf16(
            false, ap[mt], false, bv[t], (short)0, acc[mt][t], false, false);
    // LDS ops from one wave are in-order -> next iteration's writes are WAR-safe.
  }

  // ---- final normalization by l_i and store ----
  st[wid][lane] = l_i;
  wave_lds_sync();
  #pragma unroll
  for (int mt = 0; mt < 2; ++mt)
    #pragma unroll
    for (int r = 0; r < 8; ++r) {
      const float linv = 1.0f / st[wid][mt * 16 + hh * 8 + r];
      float* orow = op + (size_t)(qbase + mt * 16 + hh * 8 + r) * DD + lrow;
      #pragma unroll
      for (int t = 0; t < 4; ++t) orow[t * 16] = acc[mt][t][r] * linv;
    }
}

extern "C" void kernel_launch(void* const* d_in, const int* in_sizes, int n_in,
                              void* d_out, int out_size, void* d_ws, size_t ws_size,
                              hipStream_t stream) {
  (void)in_sizes; (void)n_in; (void)d_ws; (void)ws_size; (void)out_size;
  const float* q = (const float*)d_in[0];
  const float* k = (const float*)d_in[1];
  const float* v = (const float*)d_in[2];
  // d_in[3] is the causal mask; causality is computed analytically in-kernel.
  float* out = (float*)d_out;

  // 4096 wave-tiles (one per 32 query rows per (b,h)), 8 waves per block.
  const int blocks = (BB * HH * (SS / 32)) / 8;   // 512
  fa_fwd_causal_kernel<<<blocks, 256, 0, stream>>>(q, k, v, out);
}
